// Attention_87230785782564
// MI455X (gfx1250) — matrix operand order repaired and checked by reading, not compiled, not run on hardware
//
#include <hip/hip_runtime.h>
#include <hip/hip_bf16.h>
#include <math.h>

typedef __bf16 bf16_t;
typedef __attribute__((ext_vector_type(16))) __bf16 v16bf;
typedef __attribute__((ext_vector_type(8)))  __bf16 v8bf;   // 16 bytes
typedef __attribute__((ext_vector_type(8)))  float  v8f;

// Problem constants (match reference)
static constexpr int BSZ = 2, LSEQ = 4096, EDIM = 512, HN = 8, DHD = 64, MF = 256;
static constexpr int TOK = BSZ * LSEQ;        // 8192 tokens
static constexpr int ROWS = TOK * HN;         // 65536 (token,head) rows
static constexpr int CHUNK = 64;
static constexpr int NCH = LSEQ / CHUNK;      // 64 chunks per sequence
static constexpr int BH = BSZ * HN;           // 16 independent scans
static constexpr float DNORM = 0.35355339059327373f;  // 1/64^(1/4)
static constexpr float RATIO = 0.0625f;               // 1/sqrt(256)
static constexpr float EPSF  = 1e-6f;

// ---------------- WMMA fragment helpers ----------------

__device__ inline v8f wmma_bf16(v16bf a, v16bf b, v8f c) {
  return __builtin_amdgcn_wmma_f32_16x16x32_bf16(false, a, false, b, (short)0, c,
                                                 false, false);
}

// A-matrix 16x32 bf16, K contiguous (unit stride): lane<16 row=lane K{0..7,16..23},
// lane>=16 row=lane-16 K{8..15,24..31}.  Two 16-byte loads per lane.
__device__ inline v16bf load_frag_a_k1(const bf16_t* p, int rs) {
  int lane = threadIdx.x & 31;
  int row = lane & 15;
  int kb = (lane & 16) ? 8 : 0;
  const bf16_t* pr = p + (size_t)row * rs + kb;
  v8bf lo = *(const v8bf*)(pr);        // K kb..kb+7
  v8bf hi = *(const v8bf*)(pr + 16);   // K kb+16..kb+23
  return __builtin_shufflevector(lo, hi, 0, 1, 2, 3, 4, 5, 6, 7,
                                 8, 9, 10, 11, 12, 13, 14, 15);
}

// B-matrix 32x16 bf16, K contiguous: lane<16 col=lane K{0..7,16..23}; lane>=16 col=lane-16
// K{8..15,24..31}.  Source is column-major [col][K].  Two 16-byte loads per lane.
__device__ inline v16bf load_frag_b_k1(const bf16_t* p, int cs) {
  int lane = threadIdx.x & 31;
  int col = lane & 15;
  int kb = (lane & 16) ? 8 : 0;
  const bf16_t* pc = p + (size_t)col * cs + kb;
  v8bf lo = *(const v8bf*)(pc);
  v8bf hi = *(const v8bf*)(pc + 16);
  return __builtin_shufflevector(lo, hi, 0, 1, 2, 3, 4, 5, 6, 7,
                                 8, 9, 10, 11, 12, 13, 14, 15);
}

// C/D 16x16 f32: VGPR i -> row (lane<16?0:8)+i, col = lane&15
__device__ inline void store_frag_c(float* p, int rs, int cs, v8f c) {
  int lane = threadIdx.x & 31;
  int col = lane & 15;
  int rb = (lane & 16) ? 8 : 0;
#pragma unroll
  for (int i = 0; i < 8; ++i) p[(size_t)(rb + i) * rs + (size_t)col * cs] = c[i];
}

// ---------------- generic GEMM: C[M,N] = A[M,K] * Bt[N,K]^T (+bias[N]) ----------------
// A row-major bf16, Bt column-major ([N][K]) bf16, C fp32 row-major.
// One wave computes a 16x64 strip; K-loop is double-buffered so the next step's
// fragment loads are in flight while the current step's WMMAs execute.
__global__ void gemm_bf16_kernel(const bf16_t* __restrict__ A,
                                 const bf16_t* __restrict__ Bt,
                                 const float* __restrict__ bias,
                                 float* __restrict__ C, int Mr, int N, int K) {
  int wave = (int)((blockIdx.x * blockDim.x + threadIdx.x) >> 5);
  int ntn = N >> 6;
  int nTiles = (Mr >> 4) * ntn;
  if (wave >= nTiles) return;
  int tm = wave / ntn, tn = wave % ntn;
  const bf16_t* Ab = A + (size_t)tm * 16 * K;
  const bf16_t* Bb = Bt + (size_t)(tn * 64) * K;

  v8f acc[4] = {v8f{}, v8f{}, v8f{}, v8f{}};
  v16bf a = load_frag_a_k1(Ab, K);
  v16bf b[4];
#pragma unroll
  for (int j = 0; j < 4; ++j) b[j] = load_frag_b_k1(Bb + (size_t)j * 16 * K, K);

  for (int k = 32; k < K; k += 32) {
    // prefetch next step
    v16bf an = load_frag_a_k1(Ab + k, K);
    v16bf bn[4];
#pragma unroll
    for (int j = 0; j < 4; ++j) bn[j] = load_frag_b_k1(Bb + (size_t)j * 16 * K + k, K);
    // compute current step
#pragma unroll
    for (int j = 0; j < 4; ++j) acc[j] = wmma_bf16(a, b[j], acc[j]);
    a = an;
#pragma unroll
    for (int j = 0; j < 4; ++j) b[j] = bn[j];
  }
#pragma unroll
  for (int j = 0; j < 4; ++j) acc[j] = wmma_bf16(a, b[j], acc[j]);

  float* Cb = C + (size_t)tm * 16 * N + (size_t)tn * 64;
  int lane = threadIdx.x & 31;
#pragma unroll
  for (int j = 0; j < 4; ++j) {
    v8f c = acc[j];
    if (bias) {
      float bv = bias[tn * 64 + j * 16 + (lane & 15)];
#pragma unroll
      for (int i = 0; i < 8; ++i) c[i] += bv;
    }
    store_frag_c(Cb + j * 16, N, 1, c);
  }
}

// ---------------- prep: pack weights to bf16, transposed for the GEMMs ----------------
__global__ void prep_kernel(const float* Wq, const float* Wk, const float* Wv,
                            const float* bq, const float* bk, const float* bv,
                            const float* Wo, const float* proj,
                            bf16_t* WqkvT, float* biasqkv, bf16_t* WoT, bf16_t* projB) {
  int tid = blockIdx.x * blockDim.x + threadIdx.x;
  if (tid < 1536 * 512) {           // WqkvT[n][e], write-coalesced over e
    int n = tid >> 9, e = tid & 511;
    float v = (n < 512) ? Wq[e * 512 + n]
            : (n < 1024) ? Wk[e * 512 + (n - 512)]
                         : Wv[e * 512 + (n - 1024)];
    WqkvT[(size_t)n * 512 + e] = (bf16_t)v;
  }
  if (tid < 1536)
    biasqkv[tid] = (tid < 512) ? bq[tid] : (tid < 1024 ? bk[tid - 512] : bv[tid - 1024]);
  if (tid < 512 * 512) {            // WoT[e][hd], Wo is [hd][e]
    int e = tid >> 9, hd = tid & 511;
    WoT[(size_t)e * 512 + hd] = (bf16_t)Wo[(size_t)hd * 512 + e];
  }
  if (tid < MF * DHD)               // proj [m][d] is already [N][K] col-major
    projB[tid] = (bf16_t)proj[tid];
}

__global__ void cvt_x_kernel(const float* x, bf16_t* Xb) {
  size_t i = (size_t)blockIdx.x * blockDim.x + threadIdx.x;
  if (i < (size_t)TOK * EDIM) Xb[i] = (bf16_t)x[i];
}

// ---------------- feature prep: normalize q/k, diag terms ----------------
__global__ void feat_pre_kernel(const float* __restrict__ QKV, bf16_t* Qn, bf16_t* Kn,
                                float* diag_q, float* diag_k) {
  int wave = (int)((blockIdx.x * blockDim.x + threadIdx.x) >> 5);  // row t*8+h
  int lane = threadIdx.x & 31;
  if (wave >= ROWS) return;
  int t = wave >> 3, h = wave & 7;
  const float* qrow = QKV + (size_t)t * 1536 + h * 64;
  const float* krow = qrow + 512;
  float sq = 0.f, sk = 0.f;
  for (int i = lane; i < DHD; i += 32) {
    float q = qrow[i], k = krow[i];
    sq += q * q; sk += k * k;
    Qn[(size_t)wave * DHD + i] = (bf16_t)(q * DNORM);
    Kn[(size_t)wave * DHD + i] = (bf16_t)(k * DNORM);
  }
  for (int off = 16; off > 0; off >>= 1) {
    sq += __shfl_xor(sq, off);
    sk += __shfl_xor(sk, off);
  }
  if (lane == 0) {
    diag_q[wave] = 0.5f * DNORM * DNORM * sq;
    diag_k[wave] = 0.5f * DNORM * DNORM * sk;
  }
}

// ---------------- V transpose: QKV fp32 -> Vt bf16 [bh][d][l] ----------------
__global__ void vt_kernel(const float* __restrict__ QKV, bf16_t* __restrict__ Vt) {
  __shared__ float tile[64][65];
  int blk = blockIdx.x;              // bh*64 + lt
  int bh = blk >> 6, lt = blk & 63;
  int b = bh >> 3, h = bh & 7;
  int l0 = lt * 64;
  for (int idx = threadIdx.x; idx < 64 * 64; idx += 256) {
    int li = idx >> 6, d = idx & 63;   // read coalesced over d
    tile[li][d] = QKV[(size_t)(b * LSEQ + l0 + li) * 1536 + 1024 + h * 64 + d];
  }
  __syncthreads();
  for (int idx = threadIdx.x; idx < 64 * 64; idx += 256) {
    int d = idx >> 6, li = idx & 63;   // write coalesced over l
    Vt[((size_t)bh * 64 + d) * LSEQ + l0 + li] = (bf16_t)tile[li][d];
  }
}

// ---------------- Kp transpose: Kp [row][m] -> Kpt [bh][m][l] ----------------
__global__ void kpt_kernel(const bf16_t* __restrict__ Kp, bf16_t* __restrict__ Kpt) {
  __shared__ bf16_t tile[64][66];
  int blk = blockIdx.x;              // ((bh*4 + mt) * 64) + lt
  int lt = blk & 63;
  int mt = (blk >> 6) & 3;
  int bh = blk >> 8;
  int b = bh >> 3, h = bh & 7;
  int l0 = lt * 64, m0 = mt * 64;
  for (int idx = threadIdx.x; idx < 64 * 64; idx += 256) {
    int li = idx >> 6, m = idx & 63;   // read coalesced over m
    tile[li][m] = Kp[((size_t)(b * LSEQ + l0 + li) * 8 + h) * MF + m0 + m];
  }
  __syncthreads();
  for (int idx = threadIdx.x; idx < 64 * 64; idx += 256) {
    int m = idx >> 6, li = idx & 63;   // write coalesced over l
    Kpt[((size_t)bh * MF + m0 + m) * LSEQ + l0 + li] = tile[li][m];
  }
}

// ---------------- finish q features: per-row max stabilizer ----------------
__global__ void finish_q_kernel(const float* __restrict__ Qdash,
                                const float* __restrict__ diag_q, bf16_t* Qp) {
  int wave = (int)((blockIdx.x * blockDim.x + threadIdx.x) >> 5);
  int lane = threadIdx.x & 31;
  if (wave >= ROWS) return;
  const float* row = Qdash + (size_t)wave * MF;
  float vals[8], mx = -3.4e38f;
#pragma unroll
  for (int i = 0; i < 8; ++i) { vals[i] = row[lane * 8 + i]; mx = fmaxf(mx, vals[i]); }
  for (int off = 16; off > 0; off >>= 1) mx = fmaxf(mx, __shfl_xor(mx, off));
  float d = diag_q[wave];
#pragma unroll
  for (int i = 0; i < 8; ++i)
    Qp[(size_t)wave * MF + lane * 8 + i] =
        (bf16_t)(RATIO * (expf(vals[i] - d - mx) + EPSF));
}

// ---------------- global key max (two pass) ----------------
__global__ void kmax_pass1(const float* __restrict__ Kdash, float* partial) {
  __shared__ float sm[256];
  float mx = -3.4e38f;
  for (size_t i = (size_t)blockIdx.x * 256 + threadIdx.x; i < (size_t)ROWS * MF;
       i += (size_t)gridDim.x * 256)
    mx = fmaxf(mx, Kdash[i]);
  sm[threadIdx.x] = mx;
  __syncthreads();
  for (int s = 128; s > 0; s >>= 1) {
    if ((int)threadIdx.x < s) sm[threadIdx.x] = fmaxf(sm[threadIdx.x], sm[threadIdx.x + s]);
    __syncthreads();
  }
  if (threadIdx.x == 0) partial[blockIdx.x] = sm[0];
}
__global__ void kmax_pass2(const float* partial, float* kmax) {
  __shared__ float sm[256];
  float mx = -3.4e38f;
  for (int i = threadIdx.x; i < 1024; i += 256) mx = fmaxf(mx, partial[i]);
  sm[threadIdx.x] = mx;
  __syncthreads();
  for (int s = 128; s > 0; s >>= 1) {
    if ((int)threadIdx.x < s) sm[threadIdx.x] = fmaxf(sm[threadIdx.x], sm[threadIdx.x + s]);
    __syncthreads();
  }
  if (threadIdx.x == 0) kmax[0] = sm[0];
}

__global__ void finish_k_kernel(const float* __restrict__ Kdash,
                                const float* __restrict__ diag_k,
                                const float* __restrict__ kmax, bf16_t* Kp) {
  int wave = (int)((blockIdx.x * blockDim.x + threadIdx.x) >> 5);
  int lane = threadIdx.x & 31;
  if (wave >= ROWS) return;
  const float* row = Kdash + (size_t)wave * MF;
  float d = diag_k[wave], stab = kmax[0];
#pragma unroll
  for (int i = 0; i < 8; ++i)
    Kp[(size_t)wave * MF + lane * 8 + i] =
        (bf16_t)(RATIO * (expf(row[lane * 8 + i] - d - stab) + EPSF));
}

// ---------------- chunk state: KVt = V_c^T K_c : [64 d][256 m] per (bh,chunk) ----------------
__global__ void chunk_kv_kernel(const bf16_t* __restrict__ Vt,
                                const bf16_t* __restrict__ Kpt,
                                float* __restrict__ KVt) {
  int blk = blockIdx.x;              // bh*64 + c
  int bh = blk >> 6, c = blk & 63;
  int l0 = c * CHUNK;
  int wave = threadIdx.x >> 5;       // 16 waves
  int tm = wave >> 2, tn = wave & 3;
  const bf16_t* Ap = Vt + ((size_t)bh * 64 + tm * 16) * LSEQ + l0;   // rows=d, K=l
  const bf16_t* Bp0 = Kpt + ((size_t)bh * MF + tn * 64) * LSEQ + l0;

  // K=64: fully software-pipelined two steps
  v16bf a0 = load_frag_a_k1(Ap, LSEQ);
  v16bf b0[4];
#pragma unroll
  for (int j = 0; j < 4; ++j) b0[j] = load_frag_b_k1(Bp0 + (size_t)j * 16 * LSEQ, LSEQ);
  v16bf a1 = load_frag_a_k1(Ap + 32, LSEQ);
  v16bf b1[4];
#pragma unroll
  for (int j = 0; j < 4; ++j)
    b1[j] = load_frag_b_k1(Bp0 + (size_t)j * 16 * LSEQ + 32, LSEQ);

  v8f acc[4] = {v8f{}, v8f{}, v8f{}, v8f{}};
#pragma unroll
  for (int j = 0; j < 4; ++j) acc[j] = wmma_bf16(a0, b0[j], acc[j]);
#pragma unroll
  for (int j = 0; j < 4; ++j) acc[j] = wmma_bf16(a1, b1[j], acc[j]);

  float* Cp = KVt + (size_t)blk * (MF * DHD) + (size_t)tm * 16 * MF + tn * 64;
#pragma unroll
  for (int j = 0; j < 4; ++j) store_frag_c(Cp + j * 16, MF, 1, acc[j]);
}

// ---------------- per-chunk key sums ----------------
__global__ void ksum_kernel(const bf16_t* __restrict__ Kpt, float* __restrict__ ksum) {
  int idx = blockIdx.x * blockDim.x + threadIdx.x;   // (bh*64+c)*256 + m
  if (idx >= BH * NCH * MF) return;
  int m = idx & 255, c = (idx >> 8) & 63, bh = idx >> 14;
  const bf16_t* p = Kpt + ((size_t)bh * MF + m) * LSEQ + c * CHUNK;
  float s = 0.f;
  for (int i = 0; i < CHUNK; ++i) s += (float)p[i];
  ksum[idx] = s;
}

// ---------------- exclusive prefix over chunks (per bh) ----------------
__global__ void scan_kernel(const float* __restrict__ KVt, bf16_t* __restrict__ SpreT,
                            const float* __restrict__ ksum, float* __restrict__ kpre) {
  int bh = blockIdx.x, tid = threadIdx.x;
  for (int e0 = 0; e0 < 64; ++e0) {
    int elem = e0 * 256 + tid;       // 16384 state elements ([d][m])
    float run = 0.f;
    for (int c = 0; c < NCH; ++c) {
      size_t off = ((size_t)(bh * NCH + c)) * (MF * DHD) + elem;
      SpreT[off] = (bf16_t)run;
      run += KVt[off];
    }
  }
  {
    float run = 0.f;
    for (int c = 0; c < NCH; ++c) {
      int off = (bh * NCH + c) * MF + tid;
      kpre[off] = run;
      run += ksum[off];
    }
  }
}

// ---------------- per-chunk output ----------------
__global__ void chunk_out_kernel(const bf16_t* __restrict__ Qp,
                                 const bf16_t* __restrict__ Kp,
                                 const bf16_t* __restrict__ Vt,
                                 const bf16_t* __restrict__ SpreT,
                                 const float* __restrict__ kpre,
                                 bf16_t* __restrict__ avb) {
  __shared__ __align__(16) bf16_t Abf[CHUNK * CHUNK];
  __shared__ float denL[CHUNK];
  int blk = blockIdx.x;
  int bh = blk >> 6, c = blk & 63;
  int b = bh >> 3, h = bh & 7;
  int tb = b * LSEQ + c * CHUNK;     // global token base
  int l0 = c * CHUNK;                // in-sequence base
  int wave = threadIdx.x >> 5;       // 8 waves, 2 tiles each
  int lane = threadIdx.x & 31;
  int colL = lane & 15, rb = (lane & 16) ? 8 : 0;

  // Phase 1: Ablk = Qp_c Kp_c^T (K=256 features), causal mask, stage bf16 in LDS
#pragma unroll
  for (int tt = 0; tt < 2; ++tt) {
    int t16 = wave + tt * 8;
    int tm = t16 >> 2, tn = t16 & 3;
    const bf16_t* Ap = Qp + ((size_t)(tb + tm * 16) * 8 + h) * MF;
    const bf16_t* Bp = Kp + ((size_t)(tb + tn * 16) * 8 + h) * MF;
    v8f acc = {};
    v16bf a = load_frag_a_k1(Ap, 8 * MF);
    v16bf bfr = load_frag_b_k1(Bp, 8 * MF);
    for (int m = 32; m < MF; m += 32) {
      v16bf an = load_frag_a_k1(Ap + m, 8 * MF);
      v16bf bfn = load_frag_b_k1(Bp + m, 8 * MF);
      acc = wmma_bf16(a, bfr, acc);
      a = an; bfr = bfn;
    }
    acc = wmma_bf16(a, bfr, acc);
#pragma unroll
    for (int i = 0; i < 8; ++i) {
      int r = tm * 16 + rb + i, cj = tn * 16 + colL;
      float v = (cj <= r) ? acc[i] : 0.f;
      Abf[r * 64 + cj] = (bf16_t)v;
    }
  }
  __syncthreads();

  // Denominator: intra causal rowsum + q . ksum_prefix
  if (threadIdx.x < CHUNK) {
    int r = threadIdx.x;
    float s = 0.f;
    for (int j = 0; j <= r; ++j) s += (float)Abf[r * 64 + j];
    const bf16_t* qrow = Qp + ((size_t)(tb + r) * 8 + h) * MF;
    const float* kp = kpre + (bh * NCH + c) * MF;
    for (int m = 0; m < MF; ++m) s += (float)qrow[m] * kp[m];
    denL[r] = s;
  }
  __syncthreads();

  // Phase 2: num = Abf @ V_c + Qp_c @ S_prefix ; av = num/den
#pragma unroll
  for (int tt = 0; tt < 2; ++tt) {
    int t16 = wave + tt * 8;
    int tm = t16 >> 2, tn = t16 & 3;
    v8f acc = {};
    // intra (K = 64 chunk positions); B = V_c from Vt (K=token contiguous)
    {
      const bf16_t* Bp = Vt + ((size_t)bh * 64 + tn * 16) * LSEQ + l0;
      v16bf a0 = load_frag_a_k1(&Abf[tm * 16 * 64], 64);
      v16bf bv0 = load_frag_b_k1(Bp, LSEQ);
      v16bf a1 = load_frag_a_k1(&Abf[tm * 16 * 64 + 32], 64);
      v16bf bv1 = load_frag_b_k1(Bp + 32, LSEQ);
      acc = wmma_bf16(a0, bv0, acc);
      acc = wmma_bf16(a1, bv1, acc);
    }
    // inter (K = 256 features); B = S_prefix^T [d][m]
    const bf16_t* Ap = Qp + ((size_t)(tb + tm * 16) * 8 + h) * MF;
    const bf16_t* Bp = SpreT + (size_t)(bh * NCH + c) * (MF * DHD) + (size_t)(tn * 16) * MF;
    {
      v16bf a = load_frag_a_k1(Ap, 8 * MF);
      v16bf bfr = load_frag_b_k1(Bp, MF);
      for (int m = 32; m < MF; m += 32) {
        v16bf an = load_frag_a_k1(Ap + m, 8 * MF);
        v16bf bfn = load_frag_b_k1(Bp + m, MF);
        acc = wmma_bf16(a, bfr, acc);
        a = an; bfr = bfn;
      }
      acc = wmma_bf16(a, bfr, acc);
    }
#pragma unroll
    for (int i = 0; i < 8; ++i) {
      int r = tm * 16 + rb + i;
      float av = acc[i] / denL[r];
      avb[(size_t)(tb + r) * 512 + h * 64 + tn * 16 + colL] = (bf16_t)av;
    }
  }
}

// ---------------- launch ----------------
extern "C" void kernel_launch(void* const* d_in, const int* in_sizes, int n_in,
                              void* d_out, int out_size, void* d_ws, size_t ws_size,
                              hipStream_t stream) {
  (void)in_sizes; (void)n_in; (void)out_size; (void)ws_size;
  const float* x    = (const float*)d_in[0];
  const float* Wq   = (const float*)d_in[1];
  const float* bq   = (const float*)d_in[2];
  const float* Wk   = (const float*)d_in[3];
  const float* bk   = (const float*)d_in[4];
  const float* Wv   = (const float*)d_in[5];
  const float* bv   = (const float*)d_in[6];
  const float* Wo   = (const float*)d_in[7];
  const float* bo   = (const float*)d_in[8];
  const float* proj = (const float*)d_in[9];
  float* out = (float*)d_out;

  char* ws = (char*)d_ws;
  size_t cur = 0;
  auto alloc = [&](size_t bytes) -> char* {
    char* p = ws + cur;
    cur += (bytes + 255) & ~(size_t)255;
    return p;
  };
  bf16_t* Xb      = (bf16_t*)alloc((size_t)TOK * EDIM * 2);
  bf16_t* WqkvT   = (bf16_t*)alloc((size_t)1536 * EDIM * 2);
  float*  biasqkv = (float*) alloc(1536 * 4);
  bf16_t* WoT     = (bf16_t*)alloc((size_t)EDIM * EDIM * 2);
  bf16_t* projB   = (bf16_t*)alloc((size_t)MF * DHD * 2);
  float*  QKV     = (float*) alloc((size_t)TOK * 1536 * 4);
  bf16_t* Qn      = (bf16_t*)alloc((size_t)ROWS * DHD * 2);
  bf16_t* Kn      = (bf16_t*)alloc((size_t)ROWS * DHD * 2);
  bf16_t* Vt      = (bf16_t*)alloc((size_t)BH * DHD * LSEQ * 2);
  float*  diag_q  = (float*) alloc((size_t)ROWS * 4);
  float*  diag_k  = (float*) alloc((size_t)ROWS * 4);
  float*  Qdash   = (float*) alloc((size_t)ROWS * MF * 4);
  float*  Kdash   = (float*) alloc((size_t)ROWS * MF * 4);
  bf16_t* Qp      = (bf16_t*)alloc((size_t)ROWS * MF * 2);
  bf16_t* Kp      = (bf16_t*)alloc((size_t)ROWS * MF * 2);
  bf16_t* Kpt     = (bf16_t*)alloc((size_t)BH * MF * LSEQ * 2);
  float*  KVt     = (float*) alloc((size_t)BH * NCH * MF * DHD * 4);
  bf16_t* SpreT   = (bf16_t*)alloc((size_t)BH * NCH * MF * DHD * 2);
  float*  ksum    = (float*) alloc((size_t)BH * NCH * MF * 4);
  float*  kpre    = (float*) alloc((size_t)BH * NCH * MF * 4);
  float*  partial = (float*) alloc(1024 * 4);
  float*  kmax    = (float*) alloc(256);
  bf16_t* avb     = (bf16_t*)alloc((size_t)TOK * EDIM * 2);

  // 1. pack weights / convert inputs
  prep_kernel<<<(1536 * 512 + 255) / 256, 256, 0, stream>>>(
      Wq, Wk, Wv, bq, bk, bv, Wo, proj, WqkvT, biasqkv, WoT, projB);
  cvt_x_kernel<<<(TOK * EDIM + 255) / 256, 256, 0, stream>>>(x, Xb);

  // 2. QKV = Xb @ Wqkv + bias   [8192,1536]
  {
    int waves = (TOK / 16) * (1536 / 64);
    gemm_bf16_kernel<<<(waves * 32 + 255) / 256, 256, 0, stream>>>(
        Xb, WqkvT, biasqkv, QKV, TOK, 1536, EDIM);
  }
  // 3. feature prep + V transpose
  feat_pre_kernel<<<(ROWS * 32 + 255) / 256, 256, 0, stream>>>(
      QKV, Qn, Kn, diag_q, diag_k);
  vt_kernel<<<BH * (LSEQ / 64), 256, 0, stream>>>(QKV, Vt);

  // 4. Qdash/Kdash = [65536,64] @ proj^T[64,256]  (projB is [m][d])
  {
    int waves = (ROWS / 16) * (MF / 64);
    gemm_bf16_kernel<<<(waves * 32 + 255) / 256, 256, 0, stream>>>(
        Qn, projB, nullptr, Qdash, ROWS, MF, DHD);
    gemm_bf16_kernel<<<(waves * 32 + 255) / 256, 256, 0, stream>>>(
        Kn, projB, nullptr, Kdash, ROWS, MF, DHD);
  }
  // 5. phi(q): per-row max stabilizer
  finish_q_kernel<<<(ROWS * 32 + 255) / 256, 256, 0, stream>>>(Qdash, diag_q, Qp);
  // 6. phi(k): global max stabilizer
  kmax_pass1<<<1024, 256, 0, stream>>>(Kdash, partial);
  kmax_pass2<<<1, 256, 0, stream>>>(partial, kmax);
  finish_k_kernel<<<(ROWS * 32 + 255) / 256, 256, 0, stream>>>(Kdash, diag_k, kmax, Kp);
  kpt_kernel<<<BH * 4 * (LSEQ / 64), 256, 0, stream>>>(Kp, Kpt);

  // 7. chunked causal linear attention
  chunk_kv_kernel<<<BH * NCH, 512, 0, stream>>>(Vt, Kpt, KVt);
  ksum_kernel<<<(BH * NCH * MF + 255) / 256, 256, 0, stream>>>(Kpt, ksum);
  scan_kernel<<<BH, 256, 0, stream>>>(KVt, SpreT, ksum, kpre);
  chunk_out_kernel<<<BH * NCH, 256, 0, stream>>>(Qp, Kp, Vt, SpreT, kpre, avb);

  // 8. out = avb @ WoT^T + bo   [8192,512]
  {
    int waves = (TOK / 16) * (EDIM / 64);
    gemm_bf16_kernel<<<(waves * 32 + 255) / 256, 256, 0, stream>>>(
        avb, WoT, bo, out, TOK, EDIM, EDIM);
  }
}
